// OutAggregate_38379827757545
// MI455X (gfx1250) — 16-bit and 8-bit matrix operand order repaired and checked by reading, compiled, not run on hardware
//
#include <hip/hip_runtime.h>
#include <hip/hip_bf16.h>
#include <math.h>

#define NB 64
#define NQ 900
#define NP 960
#define NC 91
#define CP 96

#define ONE8 ((unsigned char)0x38)

typedef __attribute__((ext_vector_type(16))) _Float16 v16h;
typedef __attribute__((ext_vector_type(8)))  _Float16 v8h;
typedef __attribute__((ext_vector_type(8)))  float    v8f;
typedef __attribute__((ext_vector_type(16))) int      v16i;
typedef __attribute__((ext_vector_type(8)))  int      v8i;
typedef __attribute__((ext_vector_type(4)))  int      v4i;
typedef __attribute__((ext_vector_type(2)))  int      v2i;

__device__ __forceinline__ float wred_sumf(float v) {
#pragma unroll
  for (int o = 16; o; o >>= 1) v += __shfl_xor(v, o, 32);
  return v;
}

__device__ __forceinline__ v16i load_a_fp8(const unsigned char* __restrict__ A,
                                           int row, int hh, int k0) {
  v16i a;
  const unsigned char* rp = A + (size_t)row * NP;
#pragma unroll
  for (int q = 0; q < 2; ++q) {
    const int kb = k0 + 64 * q + 8 * hh;
    const v2i x0 = *(const v2i*)(rp + kb);
    const v2i x1 = *(const v2i*)(rp + kb + 16);
    const v2i x2 = *(const v2i*)(rp + kb + 32);
    const v2i x3 = *(const v2i*)(rp + kb + 48);
    a[8 * q + 0] = x0[0]; a[8 * q + 1] = x0[1];
    a[8 * q + 2] = x1[0]; a[8 * q + 3] = x1[1];
    a[8 * q + 4] = x2[0]; a[8 * q + 5] = x2[1];
    a[8 * q + 6] = x3[0]; a[8 * q + 7] = x3[1];
  }
  return a;
}

__device__ __forceinline__ v16i load_b_fp8(const unsigned char* __restrict__ A,
                                           int k0, int lane, int jb) {
  v16i bf;
  const int hh = lane >> 4, n = lane & 15;
#pragma unroll
  for (int i = 0; i < 16; ++i) {
    unsigned int w = 0;
#pragma unroll
    for (int e = 0; e < 4; ++e) {
      const int s = 4 * i + e;
      const int k = k0 + 64 * (s >> 5) + 16 * ((s >> 3) & 3) + 8 * hh + (s & 7);
      w |= (unsigned int)A[(size_t)k * NP + jb + n] << (8 * e);
    }
    bf[i] = (int)w;
  }
  return bf;
}

__global__ void prep_kernel(const float* __restrict__ bboxes,
                            const float* __restrict__ logits,
                            float* __restrict__ xyxy, float* __restrict__ selfent,
                            _Float16* __restrict__ qh, _Float16* __restrict__ lqT,
                            _Float16* __restrict__ Vm) {
  const int b = blockIdx.y;
  const int r = blockIdx.x * blockDim.y + threadIdx.y;
  if (r >= NP) return;
  const int lane = threadIdx.x;
  float*     xy  = xyxy + (size_t)b * NP * 4;
  _Float16*  qhb = qh  + (size_t)b * NP * CP;
  _Float16*  lqb = lqT + (size_t)b * CP * NP;
  _Float16*  Vb  = Vm  + (size_t)b * NP * CP;
  if (r < NQ) {
    const float* bx = bboxes + ((size_t)b * NQ + r) * 4;
    const float cx = bx[0], cy = bx[1], w = bx[2], h = bx[3];
    if (lane == 0) {
      xy[r * 4 + 0] = cx - 0.5f * w; xy[r * 4 + 1] = cy - 0.5f * h;
      xy[r * 4 + 2] = cx + 0.5f * w; xy[r * 4 + 3] = cy + 0.5f * h;
      Vb[r * CP + 0] = (_Float16)cx; Vb[r * CP + 1] = (_Float16)cy;
      Vb[r * CP + 2] = (_Float16)w;  Vb[r * CP + 3] = (_Float16)h;
    }
    const float* lg = logits + ((size_t)b * NQ + r) * NC;
    float p[3]; float s = 0.f;
#pragma unroll
    for (int t = 0; t < 3; ++t) {
      const int c = lane + 32 * t;
      float pv = 0.f;
      if (c < NC) pv = 1.f / (1.f + expf(-lg[c]));
      p[t] = pv; s += pv;
    }
    s = wred_sumf(s);
    const float inv = 1.f / (s + 1e-9f);
    float ent = 0.f;
#pragma unroll
    for (int t = 0; t < 3; ++t) {
      const int c = lane + 32 * t;
      if (c < CP) {
        float qv = 0.f, lqv = 0.f, pv = 0.f;
        if (c < NC) { qv = p[t] * inv; lqv = logf(qv + 1e-9f); ent += qv * lqv; pv = p[t]; }
        qhb[(size_t)r * CP + c] = (_Float16)qv;
        lqb[(size_t)c * NP + r] = (_Float16)lqv;
        if (4 + c < CP) Vb[(size_t)r * CP + 4 + c] = (_Float16)pv;
      }
    }
    ent = wred_sumf(ent);
    if (lane == 0) selfent[(size_t)b * NP + r] = ent;
  } else {
    if (lane == 0) {
      xy[r * 4 + 0] = xy[r * 4 + 1] = xy[r * 4 + 2] = xy[r * 4 + 3] = 0.f;
      selfent[(size_t)b * NP + r] = 0.f;
      Vb[r * CP + 0] = Vb[r * CP + 1] = Vb[r * CP + 2] = Vb[r * CP + 3] = (_Float16)0.f;
    }
#pragma unroll
    for (int t = 0; t < 3; ++t) {
      const int c = lane + 32 * t;
      if (c < CP) {
        qhb[(size_t)r * CP + c] = (_Float16)0.f;
        lqb[(size_t)c * NP + r] = (_Float16)0.f;
        if (4 + c < CP) Vb[(size_t)r * CP + 4 + c] = (_Float16)0.f;
      }
    }
  }
}

__global__ void mask_kernel(const float* __restrict__ xyxy,
                            const float* __restrict__ selfent,
                            const _Float16* __restrict__ qh,
                            const _Float16* __restrict__ lqT,
                            unsigned char* __restrict__ adj) {
  const int b = blockIdx.y;
  const int TI = NP / 16;
  const int tile = blockIdx.x * blockDim.y + threadIdx.y;
  if (tile >= TI * TI) return;
  const int ib = (tile / TI) * 16, jb = (tile % TI) * 16;
  const int lane = threadIdx.x, m = lane & 15, hh = lane >> 4;
  const float*     xyb = xyxy + (size_t)b * NP * 4;
  const float*     seb = selfent + (size_t)b * NP;
  const _Float16*  qhb = qh  + (size_t)b * NP * CP;
  const _Float16*  lqb = lqT + (size_t)b * CP * NP;
  unsigned char*   adjb = adj + (size_t)b * NP * NP;

  v8f acc = {};
#pragma unroll
  for (int kt = 0; kt < 3; ++kt) {
    const int k0 = kt * 32;
    const v8h lo = *(const v8h*)(qhb + (size_t)(ib + m) * CP + k0 + 8 * hh);
    const v8h hi = *(const v8h*)(qhb + (size_t)(ib + m) * CP + k0 + 16 + 8 * hh);
    v16h a;
#pragma unroll
    for (int t = 0; t < 8; ++t) { a[t] = lo[t]; a[8 + t] = hi[t]; }
    v16h bb;
#pragma unroll
    for (int s = 0; s < 16; ++s)
      bb[s] = lqb[(size_t)(k0 + (s < 8 ? 8 * hh + s : 8 + 8 * hh + s)) * NP + jb + m];
    acc = __builtin_amdgcn_wmma_f32_16x16x32_f16(false, a, false, bb, (short)0, acc,
                                                 false, false);
  }
  const int j = jb + m;
  const float bx0 = xyb[j * 4 + 0], by0 = xyb[j * 4 + 1];
  const float bx1 = xyb[j * 4 + 2], by1 = xyb[j * 4 + 3];
  const float areaB = (bx1 - bx0) * (by1 - by0);
#pragma unroll
  for (int r = 0; r < 8; ++r) {
    const int i = ib + 8 * hh + r;
    const float ax0 = xyb[i * 4 + 0], ay0 = xyb[i * 4 + 1];
    const float ax1 = xyb[i * 4 + 2], ay1 = xyb[i * 4 + 3];
    const float areaA = (ax1 - ax0) * (ay1 - ay0);
    const float iw = fmaxf(fminf(ax1, bx1) - fmaxf(ax0, bx0), 0.f);
    const float ih = fmaxf(fminf(ay1, by1) - fmaxf(ay0, by0), 0.f);
    const float inter = iw * ih;
    const float uni = areaA + areaB - inter;
    const float iou = inter / (uni + 1e-9f);
    const float ew = fmaxf(fmaxf(ax1, bx1) - fminf(ax0, bx0), 0.f);
    const float eh = fmaxf(fmaxf(ay1, by1) - fminf(ay0, by0), 0.f);
    const float areaE = ew * eh;
    const float giou = iou - (areaE - uni) / (areaE + 1e-9f);
    const float kl = seb[i] - acc[r];
    adjb[(size_t)i * NP + j] = (giou > 0.9f && kl < 0.2f) ? ONE8 : 0;
  }
}

__global__ void sym_kernel(const unsigned char* __restrict__ in,
                           unsigned char* __restrict__ out) {
  const int b = blockIdx.y;
  const int idx = blockIdx.x * blockDim.x + threadIdx.x;
  const int i = idx / NP, j = idx % NP;
  const unsigned char* ib_ = in + (size_t)b * NP * NP;
  out[(size_t)b * NP * NP + idx] = (unsigned char)(ib_[idx] | ib_[(size_t)j * NP + i]);
}

__global__ void closure_kernel(const unsigned char* __restrict__ in,
                               unsigned char* __restrict__ out) {
  const int b = blockIdx.y;
  const int BT = NP / 64;
  const int blkid = blockIdx.x * blockDim.y + threadIdx.y;
  if (blkid >= BT * BT) return;
  const int ib = (blkid / BT) * 64, jb = (blkid % BT) * 64;
  const int lane = threadIdx.x, m = lane & 15, hh = lane >> 4;
  const unsigned char* A = in + (size_t)b * NP * NP;
  unsigned char* O = out + (size_t)b * NP * NP;

  v8f acc[4][4] = {};
  for (int kt = 0; kt < 7; ++kt) {
    const int k0 = kt * 128;
    v16i af[4], bf[4];
#pragma unroll
    for (int t = 0; t < 4; ++t) af[t] = load_a_fp8(A, ib + 16 * t + m, hh, k0);
#pragma unroll
    for (int t = 0; t < 4; ++t) bf[t] = load_b_fp8(A, k0, lane, jb + 16 * t);
#pragma unroll
    for (int mi = 0; mi < 4; ++mi)
#pragma unroll
      for (int nj = 0; nj < 4; ++nj)
        acc[mi][nj] = __builtin_amdgcn_wmma_f32_16x16x128_fp8_fp8(
            af[mi], bf[nj], (short)0, acc[mi][nj], false, false);
  }
  {
    const int k0 = 896;
    v8i af[4], bf[4];
#pragma unroll
    for (int t = 0; t < 4; ++t) {
      const unsigned char* rp = A + (size_t)(ib + 16 * t + m) * NP + k0 + 8 * hh;
      const v2i p0 = *(const v2i*)(rp);
      const v2i p1 = *(const v2i*)(rp + 16);
      const v2i p2 = *(const v2i*)(rp + 32);
      const v2i p3 = *(const v2i*)(rp + 48);
      af[t][0] = p0[0]; af[t][1] = p0[1]; af[t][2] = p1[0]; af[t][3] = p1[1];
      af[t][4] = p2[0]; af[t][5] = p2[1]; af[t][6] = p3[0]; af[t][7] = p3[1];
    }
#pragma unroll
    for (int t = 0; t < 4; ++t) {
#pragma unroll
      for (int i = 0; i < 8; ++i) {
        unsigned int w = 0;
#pragma unroll
        for (int e = 0; e < 4; ++e) {
          const int s = 4 * i + e;
          const int k = k0 + 16 * (s >> 3) + 8 * hh + (s & 7);
          w |= (unsigned int)A[(size_t)k * NP + jb + 16 * t + m] << (8 * e);
        }
        bf[t][i] = (int)w;
      }
    }
#pragma unroll
    for (int mi = 0; mi < 4; ++mi)
#pragma unroll
      for (int nj = 0; nj < 4; ++nj)
        acc[mi][nj] = __builtin_amdgcn_wmma_f32_16x16x64_fp8_fp8(
            af[mi], bf[nj], (short)0, acc[mi][nj], false, false);
  }
#pragma unroll
  for (int mi = 0; mi < 4; ++mi) {
#pragma unroll
    for (int r = 0; r < 8; ++r) {
      const size_t ro = (size_t)(ib + 16 * mi + 8 * hh + r) * NP;
#pragma unroll
      for (int nj = 0; nj < 4; ++nj)
        O[ro + jb + 16 * nj + m] = acc[mi][nj][r] > 0.f ? ONE8 : 0;
    }
  }
}

__global__ void keep_kernel(const unsigned char* __restrict__ adj,
                            unsigned char* __restrict__ keep8,
                            float* __restrict__ denom) {
  const int b = blockIdx.y;
  const int r = blockIdx.x * blockDim.y + threadIdx.y;
  if (r >= NP) return;
  const int lane = threadIdx.x;
  const unsigned char* row = adj + (size_t)b * NP * NP + (size_t)r * NP;
  int cnt = 0, mn = NP;
  for (int c = lane; c < NP; c += 32)
    if (row[c]) { cnt++; if (c < mn) mn = c; }
#pragma unroll
  for (int o = 16; o; o >>= 1) {
    cnt += __shfl_xor(cnt, o, 32);
    const int om = __shfl_xor(mn, o, 32);
    mn = om < mn ? om : mn;
  }
  if (lane == 0) {
    const bool kp = (mn == r);
    keep8[(size_t)b * NP + r] = kp ? 1 : 0;
    denom[(size_t)b * NP + r] = kp ? (float)cnt : 1.0f;
  }
}

__global__ void agg_kernel(const unsigned char* __restrict__ adj,
                           const unsigned char* __restrict__ keep8,
                           const float* __restrict__ denom,
                           const _Float16* __restrict__ Vm,
                           float* __restrict__ outB, float* __restrict__ outL) {
  const int b = blockIdx.y;
  const int MT = 57, NT = CP / 16;
  const int tile = blockIdx.x * blockDim.y + threadIdx.y;
  if (tile >= MT * NT) return;
  const int ib = (tile / NT) * 16, n0 = (tile % NT) * 16;
  const int lane = threadIdx.x, m = lane & 15, hh = lane >> 4;
  const unsigned char* A = adj + (size_t)b * NP * NP;
  const _Float16* Vb = Vm + (size_t)b * NP * CP;
  const int arow = ib + m;
  const _Float16 kone = keep8[(size_t)b * NP + arow] ? (_Float16)1.f : (_Float16)0.f;
  const _Float16 hz = (_Float16)0.f;

  v8f acc = {};
  for (int kt = 0; kt < 30; ++kt) {
    const int k0 = kt * 32;
    const unsigned long long w0 =
        *(const unsigned long long*)(A + (size_t)arow * NP + k0 + 8 * hh);
    const unsigned long long w1 =
        *(const unsigned long long*)(A + (size_t)arow * NP + k0 + 16 + 8 * hh);
    v16h a;
#pragma unroll
    for (int t = 0; t < 8; ++t) {
      a[t]     = ((w0 >> (8 * t)) & 0xFFull) ? kone : hz;
      a[8 + t] = ((w1 >> (8 * t)) & 0xFFull) ? kone : hz;
    }
    v16h bb;
#pragma unroll
    for (int s = 0; s < 16; ++s)
      bb[s] = Vb[(size_t)(k0 + (s < 8 ? 8 * hh + s : 8 + 8 * hh + s)) * CP + n0 + m];
    acc = __builtin_amdgcn_wmma_f32_16x16x32_f16(false, a, false, bb, (short)0, acc,
                                                 false, false);
  }
  const int n = n0 + m;
#pragma unroll
  for (int r = 0; r < 8; ++r) {
    const int i = ib + 8 * hh + r;
    if (i >= NQ) continue;
    const float v = acc[r] / denom[(size_t)b * NP + i];
    if (n < 4) {
      outB[((size_t)b * NQ + i) * 4 + n] = v;
    } else if (n < 95) {
      const float pv = fminf(fmaxf(v, 1e-6f), 1.f - 1e-6f);
      outL[((size_t)b * NQ + i) * NC + (n - 4)] = logf(pv) - log1pf(-pv);
    }
  }
}

extern "C" void kernel_launch(void* const* d_in, const int* in_sizes, int n_in,
                              void* d_out, int out_size, void* d_ws, size_t ws_size,
                              hipStream_t stream) {
  (void)in_sizes; (void)n_in; (void)out_size; (void)ws_size;
  const float* bboxes = (const float*)d_in[0];
  const float* logits = (const float*)d_in[1];
  float* outB = (float*)d_out;
  float* outL = outB + (size_t)NB * NQ * 4;

  uintptr_t p = ((uintptr_t)d_ws + 255) & ~(uintptr_t)255;
  auto carve = [&](size_t bytes) -> void* {
    void* r = (void*)p; p += (bytes + 255) & ~(size_t)255; return r;
  };
  float* xyxy      = (float*)carve((size_t)NB * NP * 4 * sizeof(float));
  float* selfent   = (float*)carve((size_t)NB * NP * sizeof(float));
  float* denom     = (float*)carve((size_t)NB * NP * sizeof(float));
  unsigned char* keep8 = (unsigned char*)carve((size_t)NB * NP);
  _Float16* qh  = (_Float16*)carve((size_t)NB * NP * CP * 2);
  _Float16* lqT = (_Float16*)carve((size_t)NB * CP * NP * 2);
  _Float16* Vm  = (_Float16*)carve((size_t)NB * NP * CP * 2);
  unsigned char* adjA = (unsigned char*)carve((size_t)NB * NP * NP);
  unsigned char* adjB = (unsigned char*)carve((size_t)NB * NP * NP);

  const dim3 blk(32, 4);
  prep_kernel<<<dim3(NP / 4, NB), blk, 0, stream>>>(bboxes, logits, xyxy, selfent,
                                                    qh, lqT, Vm);
  mask_kernel<<<dim3((60 * 60) / 4, NB), blk, 0, stream>>>(xyxy, selfent, qh, lqT, adjA);
  sym_kernel<<<dim3((NP * NP) / 256, NB), dim3(256), 0, stream>>>(adjA, adjB);

  const unsigned char* cin = adjB;
  unsigned char* cout = adjA;
  for (int it = 0; it < 11; ++it) {
    closure_kernel<<<dim3((15 * 15 + 3) / 4, NB), blk, 0, stream>>>(cin, cout);
    const unsigned char* t = cout; cout = (unsigned char*)cin; cin = t;
  }
  const unsigned char* fin = cin;

  keep_kernel<<<dim3(NP / 4, NB), blk, 0, stream>>>(fin, keep8, denom);
  agg_kernel<<<dim3((57 * 6 + 3) / 4, NB), blk, 0, stream>>>(fin, keep8, denom, Vm,
                                                             outB, outL);
}
